// TreeLSTM_29841432773234
// MI455X (gfx1250) — compile-verified, matrix operand order checked by reading, not run on hardware
//
#include <hip/hip_runtime.h>

// ---------------------------------------------------------------------------
// TreeLSTM on MI455X (gfx1250): all GEMMs via v_wmma_f32_16x16x32_bf16 and
// v_wmma_f32_16x16x4_f32 (root fc).  Only leaf nodes need W1/W_iou (the
// reference overwrites iou for internal nodes).  h stored bf16 (WMMA A-feed),
// c stored f32 (additive recurrence).  Gate nonlinearities use the hardware
// transcendental unit (v_exp/v_rcp) instead of libm tanhf.
// ---------------------------------------------------------------------------

typedef __bf16 bf16;
typedef __attribute__((ext_vector_type(16))) __bf16 v16bf;
typedef __attribute__((ext_vector_type(8)))  float  v8f;
typedef __attribute__((ext_vector_type(2)))  float  v2f;

#define BTREES   256
#define NPER     1023
#define HDIM     128
#define NLEAF_T  512      // leaves per tree
#define LEAFBASE 511      // heap index of first leaf
#define COUT     32

union AFrag { v16bf v; unsigned int u[8]; };
union DFrag { v8f  v; float f[8]; };

// fast gate math on the transcendental pipe: ~4-5 VALU ops each
__device__ __forceinline__ float fast_rcp(float x) { return __builtin_amdgcn_rcpf(x); }
__device__ __forceinline__ float sigm(float x) { return fast_rcp(1.0f + __expf(-x)); }
__device__ __forceinline__ float tanh_f(float x) {
  // tanh(x) = 1 - 2/(exp(2x)+1); saturates correctly to +/-1 for large |x|
  return 1.0f - 2.0f * fast_rcp(1.0f + __expf(2.0f * x));
}

__device__ __forceinline__ v8f wmma_bf16(v16bf a, v16bf b, v8f c) {
  // (neg_a, A, neg_b, B, c_mod, C, reuse_a, reuse_b)
  return __builtin_amdgcn_wmma_f32_16x16x32_bf16(false, a, false, b, (short)0, c, false, false);
}

// A/B 16x32 bf16 fragment for this lane from a row-major 128-wide bf16 row.
// Layout (ISA 7.12.2): lane<16 holds K {0..7,16..23}+32*kt of row (lane&15);
// lane>=16 holds K {8..15,24..31}+32*kt.  Pairs are even => dword loads.
__device__ __forceinline__ v16bf frag_bf16_row(const bf16* rowptr, int lane, int kt) {
  AFrag a;
  const unsigned int* r = (const unsigned int*)rowptr;
  const int base = ((kt << 5) + ((lane >> 4) << 3)) >> 1;   // dword index
#pragma unroll
  for (int v = 0; v < 4; ++v) {
    a.u[v]     = r[base + v];
    a.u[v + 4] = r[base + 8 + v];
  }
  return a.v;
}

__device__ __forceinline__ unsigned int pk2(float x, float y) {
  union { bf16 h[2]; unsigned int u; } t;
  t.h[0] = (bf16)x; t.h[1] = (bf16)y;
  return t.u;   // -> v_cvt_pk_bf16_f32
}

// Same fragment, converting from an f32 source row on the fly.
__device__ __forceinline__ v16bf frag_f32_row(const float* rowptr, int lane, int kt) {
  AFrag a;
  const int klo = (kt << 5) + ((lane >> 4) << 3);
#pragma unroll
  for (int v = 0; v < 4; ++v) {
    a.u[v]     = pk2(rowptr[klo + 2 * v],      rowptr[klo + 2 * v + 1]);
    a.u[v + 4] = pk2(rowptr[klo + 16 + 2 * v], rowptr[klo + 16 + 2 * v + 1]);
  }
  return a.v;
}

// packed bf16 pair add (builds the h_cat = h0 + h1 fragment)
__device__ __forceinline__ unsigned int addpk(unsigned int a, unsigned int b) {
  union { unsigned int u; bf16 h[2]; } x, y, z;
  x.u = a; y.u = b;
  z.h[0] = (bf16)((float)x.h[0] + (float)y.h[0]);
  z.h[1] = (bf16)((float)x.h[1] + (float)y.h[1]);
  return z.u;
}

// ---------------------------------------------------------------------------
__global__ void convert_bf16_kernel(const float* __restrict__ src,
                                    bf16* __restrict__ dst, int n) {
  int i = blockIdx.x * blockDim.x + threadIdx.x;
  if (i < n) dst[i] = (bf16)src[i];
}

// ---------------------------------------------------------------------------
// Leaf kernel: per wave, 16 leaves.  x = relu(feat@W1^T+b1); iou = x@W_iou^T;
// c = sig(i)*tanh(u); h = sig(o)*tanh(c).  Writes h(bf16), c(f32).
// ---------------------------------------------------------------------------
__global__ __launch_bounds__(32) void leaf_kernel(
    const float* __restrict__ feats, const bf16* __restrict__ w1,
    const float* __restrict__ b1, const bf16* __restrict__ wiou,
    const float* __restrict__ biou, bf16* __restrict__ h_arr,
    float* __restrict__ c_arr) {
  __shared__ __align__(16) bf16 xt[16 * HDIM];     // relu(x) tile, bf16
  const int lane = threadIdx.x;
  const int tile = blockIdx.x;
  const int m = lane & 15;

  // this lane's A-row (leaf tiles never cross trees: 512 % 16 == 0)
  const int lg = tile * 16 + m;
  const size_t row = (size_t)(lg >> 9) * NPER + LEAFBASE + (lg & 511);
  const float* frow = feats + row * HDIM;

  v16bf afr[4];
#pragma unroll
  for (int kt = 0; kt < 4; ++kt) afr[kt] = frag_f32_row(frow, lane, kt);

  // GEMM1: 16x128x128, relu, stage bf16 tile in LDS
#pragma unroll
  for (int nt = 0; nt < 8; ++nt) {
    DFrag acc = {};
#pragma unroll
    for (int kt = 0; kt < 4; ++kt) {
      v16bf bfr = frag_bf16_row(w1 + (size_t)(nt * 16 + m) * HDIM, lane, kt);
      acc.v = wmma_bf16(afr[kt], bfr, acc.v);
    }
    const int col = nt * 16 + m;
    const float bias = b1[col];
#pragma unroll
    for (int r = 0; r < 8; ++r) {
      float v = acc.f[r] + bias;
      v = v > 0.0f ? v : 0.0f;
      xt[(r + ((lane >> 4) << 3)) * HDIM + col] = (bf16)v;
    }
  }
  __syncthreads();

  v16bf xfr[4];
#pragma unroll
  for (int kt = 0; kt < 4; ++kt) xfr[kt] = frag_bf16_row(xt + m * HDIM, lane, kt);

  // GEMM2: 16x384x128 (i,o,u segments share N-tile index) + activations
#pragma unroll
  for (int nt = 0; nt < 8; ++nt) {
    DFrag ai = {}, ao = {}, au = {};
#pragma unroll
    for (int kt = 0; kt < 4; ++kt) {
      ai.v = wmma_bf16(xfr[kt], frag_bf16_row(wiou + (size_t)(      nt * 16 + m) * HDIM, lane, kt), ai.v);
      ao.v = wmma_bf16(xfr[kt], frag_bf16_row(wiou + (size_t)(128 + nt * 16 + m) * HDIM, lane, kt), ao.v);
      au.v = wmma_bf16(xfr[kt], frag_bf16_row(wiou + (size_t)(256 + nt * 16 + m) * HDIM, lane, kt), au.v);
    }
    const int col = nt * 16 + m;
    const float bi = biou[col], bo = biou[128 + col], bu = biou[256 + col];
#pragma unroll
    for (int r = 0; r < 8; ++r) {
      const int rr = r + ((lane >> 4) << 3);
      const int lg2 = tile * 16 + rr;
      const size_t nrow = (size_t)(lg2 >> 9) * NPER + LEAFBASE + (lg2 & 511);
      const float cv = sigm(ai.f[r] + bi) * tanh_f(au.f[r] + bu);
      const float hv = sigm(ao.f[r] + bo) * tanh_f(cv);
      h_arr[nrow * HDIM + col] = (bf16)hv;
      c_arr[nrow * HDIM + col] = cv;
    }
  }
}

// ---------------------------------------------------------------------------
// Level kernel: per wave, 16 parents at level lvl.
//  iou = (h0+h1)@U_iou^T ; f_k = sig(h_k@U_f^T + b_f) ; c = sig(i)tanh(u)+f0c0+f1c1
// ---------------------------------------------------------------------------
__global__ __launch_bounds__(32) void level_kernel(
    int lvl, const bf16* __restrict__ uiou, const float* __restrict__ biou,
    const bf16* __restrict__ uf, const float* __restrict__ bf,
    bf16* __restrict__ h_arr, float* __restrict__ c_arr) {
  const int lane = threadIdx.x;
  const int tile = blockIdx.x;
  const int n = 1 << lvl;          // parents per tree
  const int m = lane & 15;

  const int pg = tile * 16 + m;
  const int t = pg >> lvl;
  const int node = (n - 1) + (pg & (n - 1));
  const size_t c0row = (size_t)t * NPER + 2 * node + 1;

  // get the child c-rows (f32, consumed only in the epilogue) in flight early:
  // level kernels sit on the sequential dependency chain, so hide this latency
  // behind fragment builds + 160 WMMAs.  (lowers to global_prefetch_b8)
  __builtin_prefetch(c_arr + c0row * HDIM + (lane << 3), 0, 0);
  __builtin_prefetch(c_arr + (c0row + 1) * HDIM + (lane << 3), 0, 0);

  const bf16* h0 = h_arr + c0row * HDIM;
  const bf16* h1 = h_arr + (c0row + 1) * HDIM;

  v16bf a0[4], a1[4], ac[4];
#pragma unroll
  for (int kt = 0; kt < 4; ++kt) {
    a0[kt] = frag_bf16_row(h0, lane, kt);
    a1[kt] = frag_bf16_row(h1, lane, kt);
    AFrag x, y, z;
    x.v = a0[kt]; y.v = a1[kt];
#pragma unroll
    for (int v = 0; v < 8; ++v) z.u[v] = addpk(x.u[v], y.u[v]);
    ac[kt] = z.v;
  }

#pragma unroll
  for (int nt = 0; nt < 8; ++nt) {
    DFrag fi = {}, fo = {}, fu = {}, g0 = {}, g1 = {};
#pragma unroll
    for (int kt = 0; kt < 4; ++kt) {
      const size_t wr = (size_t)(nt * 16 + m) * HDIM;
      fi.v = wmma_bf16(ac[kt], frag_bf16_row(uiou + wr,                       lane, kt), fi.v);
      fo.v = wmma_bf16(ac[kt], frag_bf16_row(uiou + wr + 128 * (size_t)HDIM, lane, kt), fo.v);
      fu.v = wmma_bf16(ac[kt], frag_bf16_row(uiou + wr + 256 * (size_t)HDIM, lane, kt), fu.v);
      v16bf bf_uf = frag_bf16_row(uf + wr, lane, kt);
      g0.v = wmma_bf16(a0[kt], bf_uf, g0.v);
      g1.v = wmma_bf16(a1[kt], bf_uf, g1.v);
    }
    const int col = nt * 16 + m;
    const float bi = biou[col], bo = biou[128 + col], bu = biou[256 + col];
    const float bff = bf[col];
#pragma unroll
    for (int r = 0; r < 8; ++r) {
      const int rr = r + ((lane >> 4) << 3);
      const int pg2 = tile * 16 + rr;
      const int t2 = pg2 >> lvl;
      const int node2 = (n - 1) + (pg2 & (n - 1));
      const size_t pr2  = (size_t)t2 * NPER + node2;
      const size_t c0r2 = (size_t)t2 * NPER + 2 * node2 + 1;
      const float c0v = c_arr[c0r2 * HDIM + col];
      const float c1v = c_arr[(c0r2 + 1) * HDIM + col];
      const float cred = sigm(g0.f[r] + bff) * c0v + sigm(g1.f[r] + bff) * c1v;
      const float cv = sigm(fi.f[r] + bi) * tanh_f(fu.f[r] + bu) + cred;
      const float hv = sigm(fo.f[r] + bo) * tanh_f(cv);
      h_arr[pr2 * HDIM + col] = (bf16)hv;
      c_arr[pr2 * HDIM + col] = cv;
    }
  }
}

// ---------------------------------------------------------------------------
// Output: out = c_root @ W_out^T + b_out  (256 x 32, K=128) in full f32 via
// V_WMMA_F32_16X16X4_F32 (falls back to scalar FMA if builtin unavailable).
// ---------------------------------------------------------------------------
__global__ __launch_bounds__(32) void out_kernel(
    const float* __restrict__ c_arr, const float* __restrict__ wout,
    const float* __restrict__ bout, float* __restrict__ out) {
  const int lane = threadIdx.x;
  const int tile = blockIdx.x;                 // 16 tiles of 16 trees
#if __has_builtin(__builtin_amdgcn_wmma_f32_16x16x4_f32)
  const int m = lane & 15;
  const float* crow = c_arr + (size_t)(tile * 16 + m) * NPER * HDIM;  // root row
#pragma unroll
  for (int nt = 0; nt < 2; ++nt) {
    DFrag acc = {};
    const float* wrow = wout + (size_t)(nt * 16 + m) * HDIM;
    for (int kt = 0; kt < 32; ++kt) {
      const int kb = kt * 4 + ((lane >> 4) << 1);   // 32-bit A 16x4 layout
      v2f a; a.x = crow[kb]; a.y = crow[kb + 1];
      v2f b; b.x = wrow[kb]; b.y = wrow[kb + 1];
      acc.v = __builtin_amdgcn_wmma_f32_16x16x4_f32(false, a, false, b,
                                                    (short)0, acc.v, false, false);
    }
    const int col = nt * 16 + m;
    const float bb = bout[col];
#pragma unroll
    for (int r = 0; r < 8; ++r) {
      const int rr = r + ((lane >> 4) << 3);
      out[(size_t)(tile * 16 + rr) * COUT + col] = acc.f[r] + bb;
    }
  }
#else
  for (int idx = lane; idx < 16 * COUT; idx += 32) {
    const int r = idx / COUT, col = idx % COUT;
    const float* crow = c_arr + (size_t)(tile * 16 + r) * NPER * HDIM;
    const float* wrow = wout + (size_t)col * HDIM;
    float acc = bout[col];
    for (int k = 0; k < HDIM; ++k) acc = fmaf(crow[k], wrow[k], acc);
    out[(size_t)(tile * 16 + r) * COUT + col] = acc;
  }
#endif
}

// ---------------------------------------------------------------------------
extern "C" void kernel_launch(void* const* d_in, const int* in_sizes, int n_in,
                              void* d_out, int out_size, void* d_ws, size_t ws_size,
                              hipStream_t stream) {
  (void)in_sizes; (void)n_in; (void)out_size; (void)ws_size;
  const float* feats = (const float*)d_in[0];
  /* d_in[1] offsets: layout is fixed heap order, roots at t*NPER */
  const float* W1    = (const float*)d_in[2];
  const float* b1    = (const float*)d_in[3];
  const float* W_iou = (const float*)d_in[4];
  const float* U_iou = (const float*)d_in[5];
  const float* b_iou = (const float*)d_in[6];
  const float* U_f   = (const float*)d_in[7];
  const float* b_f   = (const float*)d_in[8];
  const float* W_out = (const float*)d_in[9];
  const float* b_out = (const float*)d_in[10];
  float* out = (float*)d_out;

  // workspace carve-up (all 256B aligned):
  //  w1b 32KB | wioub 96KB | uioub 96KB | ufb 32KB | h(bf16) 64MB | c(f32) 128MB
  char* ws = (char*)d_ws;
  size_t off = 0;
  auto carve = [&](size_t bytes) { void* p = ws + off; off += (bytes + 255) & ~(size_t)255; return p; };
  bf16*  w1b   = (bf16*) carve((size_t)HDIM * HDIM * sizeof(bf16));
  bf16*  wioub = (bf16*) carve((size_t)3 * HDIM * HDIM * sizeof(bf16));
  bf16*  uioub = (bf16*) carve((size_t)3 * HDIM * HDIM * sizeof(bf16));
  bf16*  ufb   = (bf16*) carve((size_t)HDIM * HDIM * sizeof(bf16));
  bf16*  h_arr = (bf16*) carve((size_t)BTREES * NPER * HDIM * sizeof(bf16));
  float* c_arr = (float*)carve((size_t)BTREES * NPER * HDIM * sizeof(float));

  // weight conversion f32 -> bf16
  convert_bf16_kernel<<<(HDIM * HDIM + 255) / 256, 256, 0, stream>>>(W1, w1b, HDIM * HDIM);
  convert_bf16_kernel<<<(3 * HDIM * HDIM + 255) / 256, 256, 0, stream>>>(W_iou, wioub, 3 * HDIM * HDIM);
  convert_bf16_kernel<<<(3 * HDIM * HDIM + 255) / 256, 256, 0, stream>>>(U_iou, uioub, 3 * HDIM * HDIM);
  convert_bf16_kernel<<<(HDIM * HDIM + 255) / 256, 256, 0, stream>>>(U_f, ufb, HDIM * HDIM);

  // leaves: 131072 rows -> 8192 wave-tiles
  leaf_kernel<<<(BTREES * NLEAF_T) / 16, 32, 0, stream>>>(feats, w1b, b1, wioub, b_iou, h_arr, c_arr);

  // levels toward root (sequential dependency)
  for (int lvl = 8; lvl >= 0; --lvl) {
    const int tiles = (BTREES << lvl) / 16;
    level_kernel<<<tiles, 32, 0, stream>>>(lvl, uioub, b_iou, ufb, b_f, h_arr, c_arr);
  }

  // final fc on root c values
  out_kernel<<<BTREES / 16, 32, 0, stream>>>(c_arr, W_out, b_out, out);
}
